// TriangleAttention_1898375545269
// MI455X (gfx1250) — matrix operand order repaired and checked by reading, not compiled, not run on hardware
//
#include <hip/hip_runtime.h>
#include <hip/hip_bf16.h>

typedef __attribute__((ext_vector_type(16))) _Float16 v16h;
typedef __attribute__((ext_vector_type(8)))  _Float16 v8h;
typedef __attribute__((ext_vector_type(4)))  _Float16 v4h;
typedef __attribute__((ext_vector_type(8)))  float    v8f;

#define N_SEQ  256
#define C_IN   128
#define H_HEAD 4
#define C_H    32
#define ROWS   (N_SEQ * N_SEQ)        // 65536
#define HC     (H_HEAD * C_H)         // 128
#define VT_LD  264                    // padded LDS row stride (halfs); 264*2=528 = 33*16 -> b128 aligned rows

// ---------------------------------------------------------------------------
// WMMA helpers (CDNA5 wave32, 16x16x32 f16 -> f32)
// ---------------------------------------------------------------------------
__device__ __forceinline__ v8f wmma16(v16h a, v16h b, v8f c) {
  return __builtin_amdgcn_wmma_f32_16x16x32_f16(false, a, false, b, (short)0, c,
                                                false, false);
}

// A fragment: 16(M) x 32(K) f16 tile, row-major source, row stride ld (halfs).
// ISA layout: lane m (0..15) holds K=8g..8g+7 and 16+8g..16+8g+7 for row m,
// where g = lane>>4. Two contiguous 8-half (16B) chunks per lane.
__device__ __forceinline__ v16h load_a_frag(const _Float16* base, int ld) {
  int lane = threadIdx.x & 31;
  int m = lane & 15, g = lane >> 4;
  const _Float16* p = base + (size_t)m * ld + 8 * g;
  v8h lo = *(const v8h*)(p);
  v8h hi = *(const v8h*)(p + 16);
  v16h r;
#pragma unroll
  for (int t = 0; t < 8; ++t) { r[t] = lo[t]; r[8 + t] = hi[t]; }
  return r;
}

// B fragment: 32(K) x 16(N) f16 tile read from TRANSPOSED storage bt[n][k]
// with row stride ld (halfs); bt_base points at (n0, k0).
// ISA layout: lane g*16+n holds B[k][n] for K=8g..8g+7 and 16+8g..16+8g+7.
__device__ __forceinline__ v16h load_b_frag(const _Float16* bt_base, int ld) {
  int lane = threadIdx.x & 31;
  int n = lane & 15, g = lane >> 4;
  const _Float16* p = bt_base + (size_t)n * ld + 8 * g;
  v8h lo = *(const v8h*)(p);
  v8h hi = *(const v8h*)(p + 16);
  v16h r;
#pragma unroll
  for (int t = 0; t < 8; ++t) { r[t] = lo[t]; r[8 + t] = hi[t]; }
  return r;
}

// ---------------------------------------------------------------------------
// Kernel 1: LayerNorm (fp32 stats, f16 output) + triangle bias tri[h][q][k]
// One wave32 per (i,j) row of 128 channels.
// ---------------------------------------------------------------------------
__global__ __launch_bounds__(256) void ln_tri_kernel(
    const float* __restrict__ x, const float* __restrict__ ln_w,
    const float* __restrict__ ln_b, const float* __restrict__ w_tri,
    _Float16* __restrict__ xn16, float* __restrict__ tri) {
  int row  = (blockIdx.x * blockDim.x + threadIdx.x) >> 5;
  int lane = threadIdx.x & 31;
  if (row >= ROWS) return;
  const float4 v = *(const float4*)(x + (size_t)row * C_IN + lane * 4);
  float s  = v.x + v.y + v.z + v.w;
  float ss = v.x * v.x + v.y * v.y + v.z * v.z + v.w * v.w;
#pragma unroll
  for (int m = 16; m >= 1; m >>= 1) {
    s  += __shfl_xor(s,  m, 32);
    ss += __shfl_xor(ss, m, 32);
  }
  float mu  = s * (1.0f / C_IN);
  float var = ss * (1.0f / C_IN) - mu * mu;
  float rs  = rsqrtf(var + 1e-5f);

  float xv[4] = {v.x, v.y, v.z, v.w};
  float xn[4];
  v4h out;
#pragma unroll
  for (int t = 0; t < 4; ++t) {
    int c = lane * 4 + t;
    xn[t]  = (xv[t] - mu) * rs * ln_w[c] + ln_b[c];
    out[t] = (_Float16)xn[t];
  }
  *(v4h*)(xn16 + (size_t)row * C_IN + lane * 4) = out;

  // tri[h][row] = sum_c xn[c] * w_tri[c][h]   (w_tri is [128][4] row-major)
#pragma unroll
  for (int hh = 0; hh < H_HEAD; ++hh) {
    float p = 0.f;
#pragma unroll
    for (int t = 0; t < 4; ++t) p += xn[t] * w_tri[(lane * 4 + t) * H_HEAD + hh];
#pragma unroll
    for (int m = 16; m >= 1; m >>= 1) p += __shfl_xor(p, m, 32);
    if (lane == 0) tri[(size_t)hh * ROWS + row] = p;
  }
}

// ---------------------------------------------------------------------------
// Kernel 2: convert the five 128x128 weights to f16, TRANSPOSED [out][in].
// wq gets the 1/sqrt(C_H) query scale folded in.
// ---------------------------------------------------------------------------
__global__ __launch_bounds__(256) void wcvt_kernel(
    const float* __restrict__ wq, const float* __restrict__ wk,
    const float* __restrict__ wv, const float* __restrict__ wg,
    const float* __restrict__ wo, _Float16* __restrict__ wt) {
  int m = blockIdx.y;
  const float* src = (m == 0) ? wq : (m == 1) ? wk : (m == 2) ? wv
                    : (m == 3) ? wg : wo;
  float scale = (m == 0) ? 0.17677669529663687f : 1.0f;  // 1/sqrt(32)
  int idx = blockIdx.x * blockDim.x + threadIdx.x;       // 0..16383
  int c = idx >> 7, o = idx & 127;
  wt[(size_t)m * 16384 + o * 128 + c] = (_Float16)(src[c * 128 + o] * scale);
}

// ---------------------------------------------------------------------------
// Shared GEMM core: 16(M) x 128(N) x 128(K) per wave, A preloaded, B
// double-buffered so each B load is issued one WMMA ahead of its use.
// ---------------------------------------------------------------------------
__device__ __forceinline__ void gemm_tile_16x128(
    const _Float16* __restrict__ arow,   // A at (row0, 0), ld = 128
    const _Float16* __restrict__ bt,     // transposed weights [out][in] 128x128
    v8f acc[8]) {
  v16h a[4];
#pragma unroll
  for (int ks = 0; ks < 4; ++ks) a[ks] = load_a_frag(arow + ks * 32, C_IN);

  v16h bcur = load_b_frag(bt, 128);  // (nt=0, ks=0)
#pragma unroll
  for (int ks = 0; ks < 4; ++ks) {
#pragma unroll
    for (int nt = 0; nt < 8; ++nt) {
      int nn = nt + 1, kn = ks;
      if (nn == 8) { nn = 0; ++kn; }
      v16h bnext = (kn < 4)
          ? load_b_frag(bt + (size_t)nn * 16 * 128 + kn * 32, 128)
          : bcur;
      acc[nt] = wmma16(a[ks], bcur, acc[nt]);
      bcur = bnext;
    }
  }
}

// ---------------------------------------------------------------------------
// Kernel 3: projections  [65536,128] @ [128,128] -> q/k/v (f16) and g (sigmoid f16)
// grid: (1024, 4); block: 128 (4 waves, 16 rows each).
// ---------------------------------------------------------------------------
__global__ __launch_bounds__(128) void proj_kernel(
    const _Float16* __restrict__ xn16, const _Float16* __restrict__ wt,
    const float* __restrict__ bg, _Float16* __restrict__ q16,
    _Float16* __restrict__ k16, _Float16* __restrict__ v16g,
    _Float16* __restrict__ g16) {
  int sel  = blockIdx.y;
  int wave = threadIdx.x >> 5;
  int row0 = blockIdx.x * 64 + wave * 16;

  v8f acc[8];
#pragma unroll
  for (int nt = 0; nt < 8; ++nt) acc[nt] = (v8f){0.f,0.f,0.f,0.f,0.f,0.f,0.f,0.f};
  gemm_tile_16x128(xn16 + (size_t)row0 * C_IN, wt + (size_t)sel * 16384, acc);

  _Float16* dst = (sel == 0) ? q16 : (sel == 1) ? k16 : (sel == 2) ? v16g : g16;
  int lane = threadIdx.x & 31;
  int n = lane & 15, g = lane >> 4;
#pragma unroll
  for (int nt = 0; nt < 8; ++nt) {
#pragma unroll
    for (int j = 0; j < 8; ++j) {
      float v = acc[nt][j];
      if (sel == 3) v = 1.0f / (1.0f + __expf(-(v + bg[nt * 16 + n])));
      dst[(size_t)(row0 + j + 8 * g) * HC + nt * 16 + n] = (_Float16)v;
    }
  }
}

// ---------------------------------------------------------------------------
// Kernel 4: attention per (i, h, q-tile of 64). block = 4 waves.
// Each wave: 16 q rows. QK^T (16 WMMA, K-frags double-buffered) -> +mask/tri
// bias -> softmax (shfl) -> probs f16 via LDS -> A·V (8x2 WMMA, LDS frags
// double-buffered) -> fuse gate -> og f16.
// ---------------------------------------------------------------------------
__global__ __launch_bounds__(128) void attn_kernel(
    const _Float16* __restrict__ q16, const _Float16* __restrict__ k16,
    const _Float16* __restrict__ v16g, const _Float16* __restrict__ g16,
    const float* __restrict__ tri, const float* __restrict__ mask,
    _Float16* __restrict__ og16) {
  __shared__ alignas(16) _Float16 Vt[32 * VT_LD];           // V transposed [d][key]
  __shared__ alignas(16) _Float16 probs[4][16 * VT_LD];     // per-wave prob tile

  int i = blockIdx.x, h = blockIdx.y, qt = blockIdx.z;
  int wave = threadIdx.x >> 5, lane = threadIdx.x & 31;
  int q0 = qt * 64 + wave * 16;

  // Stage V[i,h] transposed into LDS: vectorized 16B reads, scatter to Vt[d][key]
  {
    const _Float16* vsrc = v16g + (size_t)i * N_SEQ * HC + h * C_H;
#pragma unroll
    for (int it = 0; it < 8; ++it) {
      int idx = threadIdx.x + it * 128;   // 0..1023 = key(256) x chunk(4)
      int key = idx >> 2, ch = idx & 3;
      v8h vv = *(const v8h*)(vsrc + (size_t)key * HC + ch * 8);
#pragma unroll
      for (int e = 0; e < 8; ++e) Vt[(ch * 8 + e) * VT_LD + key] = vv[e];
    }
  }
  __syncthreads();

  // Q fragment: 16 q-rows x 32 d  (K dim = head dim, one WMMA step)
  v16h aq = load_a_frag(q16 + (size_t)(i * N_SEQ + q0) * HC + h * C_H, HC);

  // Scores: S[16 x 256] = Q x K^T, 16 key-chunks of 16, B double-buffered
  v8f s[16];
  const _Float16* kbase = k16 + (size_t)i * N_SEQ * HC + h * C_H;
  {
    v16h bk = load_b_frag(kbase, HC);
    const v8f z = (v8f){0.f,0.f,0.f,0.f,0.f,0.f,0.f,0.f};
#pragma unroll
    for (int t = 0; t < 16; ++t) {
      v16h bn = (t < 15) ? load_b_frag(kbase + (size_t)(t + 1) * 16 * HC, HC) : bk;
      s[t] = wmma16(aq, bk, z);
      bk = bn;
    }
  }

  int n = lane & 15, g = lane >> 4;
  // Bias add in C/D layout: element (vgpr j, lane) = S[q0 + j + 8g][16t + n]
  const float* trib = tri + (size_t)h * ROWS;
#pragma unroll
  for (int t = 0; t < 16; ++t) {
    float mb = 1.0e9f * (mask[(size_t)i * N_SEQ + t * 16 + n] - 1.0f);
#pragma unroll
    for (int j = 0; j < 8; ++j)
      s[t][j] += mb + trib[(size_t)(q0 + j + 8 * g) * N_SEQ + t * 16 + n];
  }

  // Softmax over 256 keys: per-lane partials over t, then 16-lane-group reduce
  float mx[8], sum[8], inv[8];
#pragma unroll
  for (int j = 0; j < 8; ++j) mx[j] = -3.0e38f;
#pragma unroll
  for (int t = 0; t < 16; ++t)
#pragma unroll
    for (int j = 0; j < 8; ++j) mx[j] = fmaxf(mx[j], s[t][j]);
#pragma unroll
  for (int m = 1; m <= 8; m <<= 1)
#pragma unroll
    for (int j = 0; j < 8; ++j) mx[j] = fmaxf(mx[j], __shfl_xor(mx[j], m, 32));
#pragma unroll
  for (int j = 0; j < 8; ++j) sum[j] = 0.f;
#pragma unroll
  for (int t = 0; t < 16; ++t)
#pragma unroll
    for (int j = 0; j < 8; ++j) {
      float e = __expf(s[t][j] - mx[j]);
      s[t][j] = e;
      sum[j] += e;
    }
#pragma unroll
  for (int m = 1; m <= 8; m <<= 1)
#pragma unroll
    for (int j = 0; j < 8; ++j) sum[j] += __shfl_xor(sum[j], m, 32);
#pragma unroll
  for (int j = 0; j < 8; ++j) inv[j] = 1.0f / sum[j];

  // Normalized probabilities -> f16 in this wave's LDS tile (A-frag source)
  _Float16* pb = &probs[wave][0];
#pragma unroll
  for (int t = 0; t < 16; ++t)
#pragma unroll
    for (int j = 0; j < 8; ++j)
      pb[(j + 8 * g) * VT_LD + t * 16 + n] = (_Float16)(s[t][j] * inv[j]);
  asm volatile("" ::: "memory");  // keep prob stores before prob loads

  // O[16 x 32] = P[16 x 256] x V[256 x 32]; LDS fragments double-buffered
  v8f o0 = (v8f){0.f,0.f,0.f,0.f,0.f,0.f,0.f,0.f};
  v8f o1 = o0;
  {
    v16h ap = load_a_frag(pb, VT_LD);
    v16h b0 = load_b_frag(&Vt[0], VT_LD);
    v16h b1 = load_b_frag(&Vt[16 * VT_LD], VT_LD);
#pragma unroll
    for (int kk = 0; kk < 8; ++kk) {
      v16h apn = ap, b0n = b0, b1n = b1;
      if (kk < 7) {
        apn = load_a_frag(pb + (kk + 1) * 32, VT_LD);
        b0n = load_b_frag(&Vt[0]          + (kk + 1) * 32, VT_LD);
        b1n = load_b_frag(&Vt[16 * VT_LD] + (kk + 1) * 32, VT_LD);
      }
      o0 = wmma16(ap, b0, o0);
      o1 = wmma16(ap, b1, o1);
      ap = apn; b0 = b0n; b1 = b1n;
    }
  }

  // Fused gating: og[i, q, h*32+d] = O * sigmoid-gate
#pragma unroll
  for (int nd = 0; nd < 2; ++nd) {
    v8f o = nd ? o1 : o0;
#pragma unroll
    for (int j = 0; j < 8; ++j) {
      size_t ridx = (size_t)(i * N_SEQ + q0 + j + 8 * g) * HC + h * C_H + nd * 16 + n;
      float gv = (float)g16[ridx];
      og16[ridx] = (_Float16)(o[j] * gv);
    }
  }
}

// ---------------------------------------------------------------------------
// Kernel 5: output projection  out = og @ wo + bo   (f32 result)
// ---------------------------------------------------------------------------
__global__ __launch_bounds__(128) void out_kernel(
    const _Float16* __restrict__ og16, const _Float16* __restrict__ wt,
    const float* __restrict__ bo, float* __restrict__ out) {
  int wave = threadIdx.x >> 5;
  int row0 = blockIdx.x * 64 + wave * 16;

  v8f acc[8];
#pragma unroll
  for (int nt = 0; nt < 8; ++nt) acc[nt] = (v8f){0.f,0.f,0.f,0.f,0.f,0.f,0.f,0.f};
  gemm_tile_16x128(og16 + (size_t)row0 * HC, wt + (size_t)4 * 16384, acc);

  int lane = threadIdx.x & 31;
  int n = lane & 15, g = lane >> 4;
#pragma unroll
  for (int nt = 0; nt < 8; ++nt)
#pragma unroll
    for (int j = 0; j < 8; ++j)
      out[(size_t)(row0 + j + 8 * g) * C_IN + nt * 16 + n] =
          acc[nt][j] + bo[nt * 16 + n];
}

// ---------------------------------------------------------------------------
// Host launcher
// ---------------------------------------------------------------------------
extern "C" void kernel_launch(void* const* d_in, const int* in_sizes, int n_in,
                              void* d_out, int out_size, void* d_ws, size_t ws_size,
                              hipStream_t stream) {
  const float* x     = (const float*)d_in[0];
  const float* mask  = (const float*)d_in[1];
  const float* ln_w  = (const float*)d_in[2];
  const float* ln_b  = (const float*)d_in[3];
  const float* w_tri = (const float*)d_in[4];
  const float* wq    = (const float*)d_in[5];
  const float* wk    = (const float*)d_in[6];
  const float* wv    = (const float*)d_in[7];
  const float* wg    = (const float*)d_in[8];
  const float* bg    = (const float*)d_in[9];
  const float* wo    = (const float*)d_in[10];
  const float* bo    = (const float*)d_in[11];
  float* out = (float*)d_out;

  // Workspace carve-up (all offsets 16B aligned)
  char* ws = (char*)d_ws;
  const size_t F16MAT = (size_t)ROWS * HC * sizeof(_Float16);  // 16 MB
  _Float16* xn16 = (_Float16*)(ws);
  _Float16* q16  = (_Float16*)(ws + 1 * F16MAT);
  _Float16* k16  = (_Float16*)(ws + 2 * F16MAT);
  _Float16* v16g = (_Float16*)(ws + 3 * F16MAT);
  _Float16* g16  = (_Float16*)(ws + 4 * F16MAT);
  _Float16* og16 = (_Float16*)(ws + 5 * F16MAT);
  float*    tri  = (float*)   (ws + 6 * F16MAT);                       // 1 MB
  _Float16* wt   = (_Float16*)(ws + 6 * F16MAT + (size_t)H_HEAD * ROWS * sizeof(float));

  // 1. LayerNorm + tri bias (one wave per row)
  ln_tri_kernel<<<ROWS / 8, 256, 0, stream>>>(x, ln_w, ln_b, w_tri, xn16, tri);
  // 2. Weight convert + transpose (independent of 1; same stream serializes)
  wcvt_kernel<<<dim3(64, 5), 256, 0, stream>>>(wq, wk, wv, wg, wo, wt);
  // 3. q/k/v/g projections
  proj_kernel<<<dim3(ROWS / 64, 4), 128, 0, stream>>>(xn16, wt, bg, q16, k16, v16g, g16);
  // 4. attention + gating
  attn_kernel<<<dim3(N_SEQ, H_HEAD, 4), 128, 0, stream>>>(q16, k16, v16g, g16, tri, mask, og16);
  // 5. output projection
  out_kernel<<<ROWS / 64, 128, 0, stream>>>(og16, wt, bo, out);
}
